// SinkhornLoss_27049704030884
// MI455X (gfx1250) — compile-verified, matrix operand order checked by reading, not run on hardware
//
#include <hip/hip_runtime.h>

typedef __attribute__((ext_vector_type(16))) _Float16 v16h;
typedef __attribute__((ext_vector_type(8)))  float    v8f;

#define BATCH   8
#define NROW    2048
#define MCOL    2048
#define NITERS  20
#define NCHUNK  16         // row chunks for column pass (128 rows each)

// Work in base-2 log domain: pot2 = pot * log2(e).
// exp(-D/eps - pot) == exp2(-D*IE2 - pot2), IE2 = log2(e)/eps
#define IE2 14.426950408889634f   // (1/0.1) * log2(e)

__device__ __forceinline__ float fast_exp2(float x) { return __builtin_amdgcn_exp2f(x); }
__device__ __forceinline__ float fast_log2(float x) { return __builtin_amdgcn_logf(x); }

// ---------------------------------------------------------------- zero init
__global__ void k_zero(float* __restrict__ p, int n) {
    int i = blockIdx.x * blockDim.x + threadIdx.x;
    if (i < n) p[i] = 0.0f;
}

// ---------------------------------------------------------------- row pass
// f2_i = log2 sum_j exp2(-D_ij*IE2 - g2_j)
// One wave32 per 16-row tile. exp2 on the trans pipe (1 VALU fma + 1 TRANS
// per element, co-executing), row-sum on the matrix pipe via
// v_wmma_f32_16x16x32_f16 against a ones B-matrix (f32 accumulate).
__global__ __launch_bounds__(32) void k_rowpass(const float* __restrict__ D,
                                                const float* __restrict__ g2,
                                                float* __restrict__ f2) {
    const int tile = blockIdx.x;            // 0 .. BATCH*(NROW/16)-1
    const int b    = tile >> 7;             // 128 tiles per batch
    const int row0 = (tile & 127) << 4;
    const float* __restrict__ Db = D  + (size_t)b * NROW * MCOL;
    const float* __restrict__ gb = g2 + (size_t)b * MCOL;

    const int lane = threadIdx.x;           // 0..31
    const int m    = lane & 15;             // row within tile (A-matrix M)
    const int kg   = (lane >> 4) << 3;      // lanes 0-15: K 0-7/16-23; 16-31: K 8-15/24-31
    const float* __restrict__ rowPtr = Db + (size_t)(row0 + m) * MCOL;

    v16h ones;
#pragma unroll
    for (int t = 0; t < 16; ++t) ones[t] = (_Float16)1.0f;

    const float nie = -IE2;
    v8f acc = {};                           // f32 C/D accumulator (row sums)

    for (int k0 = 0; k0 < MCOL; k0 += 32) {
        __builtin_prefetch(rowPtr + k0 + 512, 0, 1);  // global_prefetch_b8

        const float4* dv = reinterpret_cast<const float4*>(rowPtr + k0 + kg);
        const float4* gv = reinterpret_cast<const float4*>(gb     + k0 + kg);
        float4 d0 = dv[0], d1 = dv[1], d2 = dv[4], d3 = dv[5]; // +0..7, +16..23
        float4 g0 = gv[0], g1 = gv[1], g2v = gv[4], g3 = gv[5];

        float ev[16] = {
            fast_exp2(fmaf(d0.x, nie, -g0.x)),  fast_exp2(fmaf(d0.y, nie, -g0.y)),
            fast_exp2(fmaf(d0.z, nie, -g0.z)),  fast_exp2(fmaf(d0.w, nie, -g0.w)),
            fast_exp2(fmaf(d1.x, nie, -g1.x)),  fast_exp2(fmaf(d1.y, nie, -g1.y)),
            fast_exp2(fmaf(d1.z, nie, -g1.z)),  fast_exp2(fmaf(d1.w, nie, -g1.w)),
            fast_exp2(fmaf(d2.x, nie, -g2v.x)), fast_exp2(fmaf(d2.y, nie, -g2v.y)),
            fast_exp2(fmaf(d2.z, nie, -g2v.z)), fast_exp2(fmaf(d2.w, nie, -g2v.w)),
            fast_exp2(fmaf(d3.x, nie, -g3.x)),  fast_exp2(fmaf(d3.y, nie, -g3.y)),
            fast_exp2(fmaf(d3.z, nie, -g3.z)),  fast_exp2(fmaf(d3.w, nie, -g3.w))
        };
        v16h a;
#pragma unroll
        for (int t = 0; t < 16; ++t) a[t] = (_Float16)ev[t];

        // D = A x ones + C : every N column of D holds the row sum of A
        acc = __builtin_amdgcn_wmma_f32_16x16x32_f16(
            /*neg_a=*/false, a, /*neg_b=*/false, ones,
            /*c_mod=*/(short)0, acc, /*reuse_a=*/false, /*reuse_b=*/false);
    }

    // C/D layout: lane 0 VGPR r -> row r; lane 16 VGPR r -> row r+8
    float* __restrict__ fb = f2 + (size_t)b * NROW + row0;
    if (lane == 0) {
#pragma unroll
        for (int r = 0; r < 8; ++r) fb[r] = fast_log2(acc[r]);
    } else if (lane == 16) {
#pragma unroll
        for (int r = 0; r < 8; ++r) fb[8 + r] = fast_log2(acc[r]);
    }
}

// ---------------------------------------------------------------- col pass
// partial[b][chunk][j] = sum_{i in chunk} exp2(-D_ij*IE2 - f2_i)
// block = 256 threads, each owns 4 columns (float4 coalesced loads).
__global__ __launch_bounds__(256) void k_colpass(const float* __restrict__ D,
                                                 const float* __restrict__ f2,
                                                 float* __restrict__ partial) {
    const int ct    = blockIdx.x;           // 0..1 (1024-column tiles)
    const int chunk = blockIdx.y;           // 0..NCHUNK-1
    const int b     = blockIdx.z;           // 0..BATCH-1
    const int j     = ct * 1024 + threadIdx.x * 4;
    const float* __restrict__ Db = D  + (size_t)b * NROW * MCOL;
    const float* __restrict__ fb = f2 + (size_t)b * NROW;
    const int i0 = chunk * (NROW / NCHUNK);
    const float nie = -IE2;

    float4 acc = {0.f, 0.f, 0.f, 0.f};
#pragma unroll 4
    for (int i = i0; i < i0 + NROW / NCHUNK; ++i) {
        const float fi = fb[i];             // uniform -> scalar load
        float4 d = *reinterpret_cast<const float4*>(Db + (size_t)i * MCOL + j);
        acc.x += fast_exp2(fmaf(d.x, nie, -fi));
        acc.y += fast_exp2(fmaf(d.y, nie, -fi));
        acc.z += fast_exp2(fmaf(d.z, nie, -fi));
        acc.w += fast_exp2(fmaf(d.w, nie, -fi));
    }
    *reinterpret_cast<float4*>(partial + ((size_t)(b * NCHUNK + chunk)) * MCOL + j) = acc;
}

// g2_j = log2( sum over chunks of partial ), fixed order -> deterministic
__global__ __launch_bounds__(256) void k_colfinish(const float* __restrict__ partial,
                                                   float* __restrict__ g2) {
    const int idx = blockIdx.x * blockDim.x + threadIdx.x;  // 0..BATCH*MCOL-1
    const int b = idx >> 11;
    const int j = idx & (MCOL - 1);
    float s = 0.0f;
#pragma unroll
    for (int c = 0; c < NCHUNK; ++c)
        s += partial[((size_t)(b * NCHUNK + c)) * MCOL + j];
    g2[idx] = fast_log2(s);
}

// ---------------------------------------------------------------- loss pass
// block partial of sum_ij D_ij * exp2(-D_ij*IE2 - f2_i - g2_j)
__global__ __launch_bounds__(256) void k_losspass(const float* __restrict__ D,
                                                  const float* __restrict__ f2,
                                                  const float* __restrict__ g2,
                                                  float* __restrict__ lpart) {
    const int ct    = blockIdx.x;
    const int chunk = blockIdx.y;
    const int b     = blockIdx.z;
    const int tid   = threadIdx.x;
    const int j     = ct * 1024 + tid * 4;
    const float* __restrict__ Db = D  + (size_t)b * NROW * MCOL;
    const float* __restrict__ fb = f2 + (size_t)b * NROW;
    const float4 gj = *reinterpret_cast<const float4*>(g2 + (size_t)b * MCOL + j);
    const int i0 = chunk * (NROW / NCHUNK);
    const float nie = -IE2;

    float4 acc = {0.f, 0.f, 0.f, 0.f};
#pragma unroll 4
    for (int i = i0; i < i0 + NROW / NCHUNK; ++i) {
        const float fi = fb[i];
        float4 d = *reinterpret_cast<const float4*>(Db + (size_t)i * MCOL + j);
        acc.x += d.x * fast_exp2(fmaf(d.x, nie, -fi - gj.x));
        acc.y += d.y * fast_exp2(fmaf(d.y, nie, -fi - gj.y));
        acc.z += d.z * fast_exp2(fmaf(d.z, nie, -fi - gj.z));
        acc.w += d.w * fast_exp2(fmaf(d.w, nie, -fi - gj.w));
    }
    __shared__ float red[256];
    red[tid] = (acc.x + acc.y) + (acc.z + acc.w);
    __syncthreads();
    for (int s = 128; s > 0; s >>= 1) {
        if (tid < s) red[tid] += red[tid + s];
        __syncthreads();
    }
    if (tid == 0) lpart[(b * NCHUNK + chunk) * 2 + ct] = red[0];
}

__global__ __launch_bounds__(256) void k_lossfinish(const float* __restrict__ lpart,
                                                    float* __restrict__ out) {
    const int tid = threadIdx.x;
    __shared__ float red[256];
    red[tid] = lpart[tid];                  // BATCH*NCHUNK*2 == 256 partials
    __syncthreads();
    for (int s = 128; s > 0; s >>= 1) {
        if (tid < s) red[tid] += red[tid + s];
        __syncthreads();
    }
    if (tid == 0) out[0] = red[0] * (1.0f / BATCH);
}

// ---------------------------------------------------------------- launcher
extern "C" void kernel_launch(void* const* d_in, const int* in_sizes, int n_in,
                              void* d_out, int out_size, void* d_ws, size_t ws_size,
                              hipStream_t stream) {
    (void)in_sizes; (void)n_in; (void)out_size; (void)ws_size;
    const float* D = (const float*)d_in[0];
    float* out = (float*)d_out;

    float* ws          = (float*)d_ws;
    float* f2          = ws;                          // BATCH*NROW
    float* g2          = f2 + BATCH * NROW;           // BATCH*MCOL
    float* colPartial  = g2 + BATCH * MCOL;           // BATCH*NCHUNK*MCOL
    float* lossPartial = colPartial + (size_t)BATCH * NCHUNK * MCOL;  // 256

    k_zero<<<(BATCH * MCOL + 255) / 256, 256, 0, stream>>>(g2, BATCH * MCOL);

    for (int it = 0; it < NITERS; ++it) {
        k_rowpass<<<BATCH * (NROW / 16), 32, 0, stream>>>(D, g2, f2);
        k_colpass<<<dim3(2, NCHUNK, BATCH), 256, 0, stream>>>(D, f2, colPartial);
        k_colfinish<<<(BATCH * MCOL) / 256, 256, 0, stream>>>(colPartial, g2);
    }
    k_losspass<<<dim3(2, NCHUNK, BATCH), 256, 0, stream>>>(D, f2, g2, lossPartial);
    k_lossfinish<<<1, 256, 0, stream>>>(lossPartial, out);
}
